// KA_D_71236327572007
// MI455X (gfx1250) — compile-verified, matrix operand order checked by reading, not run on hardware
//
#include <hip/hip_runtime.h>
#include <hip/hip_bf16.h>
#include <math.h>

// ---------------------------------------------------------------------------
// Problem constants (from reference)
// ---------------------------------------------------------------------------
#define BB     1000
#define WIN    15
#define WORD   300
#define HH     100
#define TRANSE 100
#define MID    200
#define TYPES  74
#define MAXC   20

// padded-K sizes (multiples of 32 halves -> 64B rows, 16B-aligned chunks)
#define KP_X   320   // WORD=300 -> 320
#define KP_H   128   // HH=100  -> 128
#define KP_Q1  512   // 500 -> 512
#define KP_Q2  224   // 200 -> 224
#define KP_W   704   // 700 -> 704
#define KP_T   128   // 100 -> 128

typedef _Float16 v16h __attribute__((ext_vector_type(16)));
typedef _Float16 v8h  __attribute__((ext_vector_type(8)));
typedef float    v8f  __attribute__((ext_vector_type(8)));
typedef unsigned int u32x4 __attribute__((ext_vector_type(4)));
typedef int i32x8 __attribute__((ext_vector_type(8)));
typedef int i32x4 __attribute__((ext_vector_type(4)));

#define AS1 __attribute__((address_space(1)))
#define AS3 __attribute__((address_space(3)))

#define A_STRIDE 40   // A tile LDS row stride in halves (80B, 16B aligned)

// ---- CDNA5 feature probes (graceful fallback if builtins absent) ----------
#if defined(__has_builtin)
#if __has_builtin(__builtin_amdgcn_global_load_async_to_lds_b128)
#define HAS_ASYNC_COPY 1
#endif
#if __has_builtin(__builtin_amdgcn_tensor_load_to_lds)
#define HAS_TDM 1
#endif
#endif

__device__ inline void wait_async0() {
#if defined(__has_builtin)
#if __has_builtin(__builtin_amdgcn_s_wait_asynccnt)
    __builtin_amdgcn_s_wait_asynccnt(0);
#else
    asm volatile("s_wait_asynccnt 0x0" ::: "memory");
#endif
#else
    asm volatile("s_wait_asynccnt 0x0" ::: "memory");
#endif
}

#ifdef HAS_ASYNC_COPY
// async 16B copy global->LDS (param types per hipcc diagnostic: int4* operands)
__device__ inline void async_cp_b128(const _Float16* g, _Float16* l) {
    __builtin_amdgcn_global_load_async_to_lds_b128(
        (AS1 i32x4*)g, (AS3 i32x4*)l, 0, 0);
}
#endif

#ifdef HAS_TDM
// 2-D TDM load: tile [tile_dim1=64 rows] x [tile_dim0=Kpad elems], 2-byte elems.
__device__ inline void tdm_load_2d(u32x4 g0, i32x8 g1) {
    i32x4 z4 = {0, 0, 0, 0};
#if __clang_major__ >= 23
    i32x8 z8 = {0, 0, 0, 0, 0, 0, 0, 0};
    __builtin_amdgcn_tensor_load_to_lds(g0, g1, z4, z4, z8, 0);
#else
    __builtin_amdgcn_tensor_load_to_lds(g0, g1, z4, z4, 0);
#endif
}
#endif

// ---------------------------------------------------------------------------
// Fragment load from LDS (ISA 16-bit operand layout):
// lane holds row/col (lane&15); K chunks [kb,kb+8) and [kb+16,kb+24),
// kb = (lane>=16)*8.  Two 16B LDS reads.
// ---------------------------------------------------------------------------
__device__ inline v16h load_frag(const _Float16* __restrict__ p, int kb) {
    v8h lo = *(const v8h*)(p + kb);
    v8h hi = *(const v8h*)(p + kb + 16);
    v16h r;
#pragma unroll
    for (int i = 0; i < 8; ++i) { r[i] = lo[i]; r[i + 8] = hi[i]; }
    return r;
}

__device__ inline void store_c(float* __restrict__ C, _Float16* __restrict__ C16,
                               int gm, int gn, int M, int N, int ldc, int ldc16,
                               float v, int beta, int act) {
    if (gm < M && gn < N) {
        float o = v;
        if (beta && C) o += C[(size_t)gm * ldc + gn];
        if (act == 1) o = tanhf(o);
        else if (act == 2) o = 1.0f / (1.0f + expf(-o));
        if (C)   C[(size_t)gm * ldc + gn] = o;
        if (C16) C16[(size_t)gm * ldc16 + gn] = (_Float16)o;
    }
}

// ---------------------------------------------------------------------------
// WMMA GEMM:  C[M,N] (f32 and/or f16) = act( A16[M,Kpad] @ Bt16[N,Kpad]^T (+C) )
// A and Bt are f16 with row stride = Kpad (multiple of 32, zero-padded).
// Block: 128 threads = 4 waves; 64x64 tile; wave -> 32x32 (2x2 WMMA frags).
// B tile staged ONCE per block (TDM if available), A tiles streamed per K step
// (async-to-LDS if available).
// ---------------------------------------------------------------------------
__global__ __launch_bounds__(128)
void wmma_gemm_kernel(const _Float16* __restrict__ Am, const _Float16* __restrict__ Bt,
                      float* __restrict__ C, _Float16* __restrict__ C16,
                      int M, int N, int Kpad, int ldc, int ldc16,
                      int beta, int act) {
    extern __shared__ __align__(16) _Float16 smem[];
    _Float16* sB = smem;                         // [64][Kpad]
    _Float16* sA = smem + (size_t)64 * Kpad;     // [64][A_STRIDE]

    const int tid  = threadIdx.x;
    const int lane = tid & 31;
    const int wave = tid >> 5;
    const int rowBase = blockIdx.y * 64;
    const int colBase = blockIdx.x * 64;
    const int wm = (wave >> 1) * 32;
    const int wn = (wave & 1) * 32;

    // ---- stage whole B tile (rows [colBase, colBase+64) x Kpad) ----
#ifdef HAS_TDM
    if (wave == 0) {
        unsigned lds_b = (unsigned)__builtin_amdgcn_groupstaticsize(); // dyn LDS base
        unsigned long long ga = (unsigned long long)(const void*)(Bt + (size_t)colBase * Kpad);
        int dim1 = N - colBase;   // rows available; beyond -> OOB reads return 0
        u32x4 g0;
        g0[0] = 1u;                                             // count=1
        g0[1] = lds_b;                                          // lds_addr
        g0[2] = (unsigned)ga;                                   // global_addr[31:0]
        g0[3] = (unsigned)((ga >> 32) & 0x1FFFFFFu) | (2u << 30); // addr[56:32] | type=2
        i32x8 g1;
        g1[0] = 0x00010000;                                     // data_size = 2 bytes
        g1[1] = (Kpad & 0xffff) << 16;                          // tensor_dim0[15:0]
        g1[2] = ((Kpad >> 16) & 0xffff) | ((dim1 & 0xffff) << 16); // dim0 hi | dim1 lo
        g1[3] = ((dim1 >> 16) & 0xffff) | ((Kpad & 0xffff) << 16); // dim1 hi | tile_dim0
        g1[4] = 64;                                             // tile_dim1 = 64
        g1[5] = Kpad;                                           // tensor_dim0_stride lo
        g1[6] = 0;
        g1[7] = 0;
        tdm_load_2d(g0, g1);
        __builtin_amdgcn_s_wait_tensorcnt(0);
    }
#elif defined(HAS_ASYNC_COPY)
    {
        int cpr = Kpad >> 3;          // 16B chunks per row
        int total = 64 * cpr;
        for (int c2 = tid; c2 < total; c2 += 128) {
            int rowb = c2 / cpr, q16 = c2 - rowb * cpr;
            int gn = colBase + rowb;
            if (gn < N)
                async_cp_b128(Bt + (size_t)gn * Kpad + q16 * 8,
                              sB + (size_t)rowb * Kpad + q16 * 8);
        }
        wait_async0();
    }
#else
    {
        int cpr = Kpad >> 3;
        int total = 64 * cpr;
        for (int c2 = tid; c2 < total; c2 += 128) {
            int rowb = c2 / cpr, q16 = c2 - rowb * cpr;
            int gn = colBase + rowb;
            v8h val = {};
            if (gn < N) val = *(const v8h*)(Bt + (size_t)gn * Kpad + q16 * 8);
            *(v8h*)(sB + (size_t)rowb * Kpad + q16 * 8) = val;
        }
    }
#endif
    __syncthreads();

    v8f acc00 = {}; v8f acc01 = {}; v8f acc10 = {}; v8f acc11 = {};

    for (int k0 = 0; k0 < Kpad; k0 += 32) {
        // ---- stage A tile [64 rows x 32 halves]: 256 x 16B chunks ----
#ifdef HAS_ASYNC_COPY
        for (int c2 = tid; c2 < 256; c2 += 128) {
            int rowa = c2 >> 2, q16 = c2 & 3;
            int gr = rowBase + rowa;
            if (gr < M)
                async_cp_b128(Am + (size_t)gr * Kpad + k0 + q16 * 8,
                              sA + rowa * A_STRIDE + q16 * 8);
        }
        wait_async0();
#else
        for (int c2 = tid; c2 < 256; c2 += 128) {
            int rowa = c2 >> 2, q16 = c2 & 3;
            int gr = rowBase + rowa;
            v8h val = {};
            if (gr < M) val = *(const v8h*)(Am + (size_t)gr * Kpad + k0 + q16 * 8);
            *(v8h*)(sA + rowa * A_STRIDE + q16 * 8) = val;
        }
#endif
        __syncthreads();

        const int kb = (lane >> 4) * 8;
        const int mr = lane & 15;
        v16h a0 = load_frag(&sA[(wm      + mr) * A_STRIDE], kb);
        v16h a1 = load_frag(&sA[(wm + 16 + mr) * A_STRIDE], kb);
        v16h b0 = load_frag(&sB[(size_t)(wn      + mr) * Kpad + k0], kb);
        v16h b1 = load_frag(&sB[(size_t)(wn + 16 + mr) * Kpad + k0], kb);

        acc00 = __builtin_amdgcn_wmma_f32_16x16x32_f16(false, a0, false, b0, (short)0, acc00, false, false);
        acc01 = __builtin_amdgcn_wmma_f32_16x16x32_f16(false, a0, false, b1, (short)0, acc01, false, false);
        acc10 = __builtin_amdgcn_wmma_f32_16x16x32_f16(false, a1, false, b0, (short)0, acc10, false, false);
        acc11 = __builtin_amdgcn_wmma_f32_16x16x32_f16(false, a1, false, b1, (short)0, acc11, false, false);
        __syncthreads();
    }

    // ---- epilogue: C/D layout: VGPR r -> M = r + (lane>=16)*8, N = lane&15 ----
    const int nn = lane & 15;
    const int mo = (lane >> 4) * 8;
#pragma unroll
    for (int r = 0; r < 8; ++r) {
        store_c(C, C16, rowBase + wm +      mo + r, colBase + wn +      nn, M, N, ldc, ldc16, acc00[r], beta, act);
        store_c(C, C16, rowBase + wm +      mo + r, colBase + wn + 16 + nn, M, N, ldc, ldc16, acc01[r], beta, act);
        store_c(C, C16, rowBase + wm + 16 + mo + r, colBase + wn +      nn, M, N, ldc, ldc16, acc10[r], beta, act);
        store_c(C, C16, rowBase + wm + 16 + mo + r, colBase + wn + 16 + nn, M, N, ldc, ldc16, acc11[r], beta, act);
    }
}

// ---------------------------------------------------------------------------
// Precompute kernels
// ---------------------------------------------------------------------------
// X16[t*4000 + r][KP_X]: stacked LSTM input in f16 (rows remapped, K padded)
__global__ void pack_x16_kernel(const float* __restrict__ left,
                                const float* __restrict__ right,
                                _Float16* __restrict__ X16) {
    long id = (long)blockIdx.x * blockDim.x + threadIdx.x;
    const long total = (long)WIN * 4 * BB * KP_X;
    if (id >= total) return;
    int col = (int)(id % KP_X);
    long row = id / KP_X;
    int t = (int)(row / (4 * BB));
    int r = (int)(row % (4 * BB));
    float v = 0.0f;
    if (col < WORD) {
        int q = r / BB, b = r - q * BB;
        const float* src;
        if (q == 0)      src = left  + ((size_t)(WIN - 1 - t) * BB + b) * WORD;
        else if (q == 1) src = right + ((size_t)(WIN - 1 - t) * BB + b) * WORD;
        else if (q == 2) src = left  + ((size_t)t * BB + b) * WORD;
        else             src = right + ((size_t)t * BB + b) * WORD;
        v = src[col];
    }
    X16[id] = (_Float16)v;
}

// dst[n][Kpad] = src[srcRow0+k][N] (transpose + cvt + zero pad)
__global__ void transpose_cvt_kernel(const float* __restrict__ src, _Float16* __restrict__ dst,
                                     int K, int N, int Kpad, int srcRow0) {
    int id = blockIdx.x * blockDim.x + threadIdx.x;
    if (id >= N * Kpad) return;
    int n = id / Kpad, k = id - n * Kpad;
    float v = (k < K) ? src[(size_t)(srcRow0 + k) * N + n] : 0.0f;
    dst[id] = (_Float16)v;
}

// dst[n][Kpad] = src[n][k] (cvt + zero pad, no transpose)
__global__ void pad_cvt_kernel(const float* __restrict__ src, _Float16* __restrict__ dst,
                               int K, int N, int Kpad) {
    int id = blockIdx.x * blockDim.x + threadIdx.x;
    if (id >= N * Kpad) return;
    int n = id / Kpad, k = id - n * Kpad;
    float v = (k < K) ? src[(size_t)n * K + k] : 0.0f;
    dst[id] = (_Float16)v;
}

__global__ void fill_zero_f_kernel(float* __restrict__ p, int n) {
    int id = blockIdx.x * blockDim.x + threadIdx.x;
    if (id < n) p[id] = 0.0f;
}
__global__ void fill_zero_h_kernel(_Float16* __restrict__ p, int n) {
    int id = blockIdx.x * blockDim.x + threadIdx.x;
    if (id < n) p[id] = (_Float16)0.0f;
}

// ---------------------------------------------------------------------------
// Elementwise LSTM cell (TF gate order i, j, f, o; forget_bias = 1.0)
// gates[n,4H] = x@Wx + h@Wh; writes c (f32), h16 (padded f16), optional hs (f32)
// ---------------------------------------------------------------------------
__global__ void lstm_cell_kernel(const float* __restrict__ gates,
                                 const float* __restrict__ bias,
                                 float* __restrict__ c, _Float16* __restrict__ h16,
                                 float* __restrict__ hs_t, int Nrows) {
    int id = blockIdx.x * blockDim.x + threadIdx.x;
    int total = Nrows * HH;
    if (id >= total) return;
    int n = id / HH, k = id - n * HH;
    const float* g = gates + (size_t)n * 4 * HH;
    float gi = g[k]          + bias[k];
    float gj = g[HH + k]     + bias[HH + k];
    float gf = g[2 * HH + k] + bias[2 * HH + k];
    float go = g[3 * HH + k] + bias[3 * HH + k];
    float cv = c[id];
    float sf = 1.0f / (1.0f + expf(-(gf + 1.0f)));
    float si = 1.0f / (1.0f + expf(-gi));
    float so = 1.0f / (1.0f + expf(-go));
    cv = sf * cv + si * tanhf(gj);
    float hv = so * tanhf(cv);
    c[id] = cv;
    h16[(size_t)n * KP_H + k] = (_Float16)hv;
    if (hs_t) hs_t[id] = hv;
}

// xq16[b][KP_Q1]: 0..299 entity, 300..399 lq=h_q[b], 400..499 rq=h_q[1000+b], pad 0
__global__ void pack_xq16_kernel(const float* __restrict__ entity,
                                 const _Float16* __restrict__ h_q,
                                 _Float16* __restrict__ xq) {
    int id = blockIdx.x * blockDim.x + threadIdx.x;
    if (id >= BB * KP_Q1) return;
    int b = id / KP_Q1, col = id - b * KP_Q1;
    _Float16 v = (_Float16)0.0f;
    if (col < WORD)                v = (_Float16)entity[(size_t)b * WORD + col];
    else if (col < WORD + HH)      v = h_q[(size_t)b * KP_H + (col - WORD)];
    else if (col < WORD + 2 * HH)  v = h_q[(size_t)(BB + b) * KP_H + (col - WORD - HH)];
    xq[id] = v;
}

// Nearest candidate + miss logic + test select -> Q16[b][KP_H] (pads pre-zeroed)
__global__ void disamb_kernel(const float* __restrict__ embedding,
                              const int* __restrict__ disamb,
                              const unsigned char* __restrict__ test,
                              const float* __restrict__ query_,
                              const float* __restrict__ query,
                              _Float16* __restrict__ Q16) {
    int b = blockIdx.x * blockDim.x + threadIdx.x;
    if (b >= BB) return;
    const float* qb = query + (size_t)b * TRANSE;
    float best = 3.4e38f;
    int bestj = 0;
    for (int j = 0; j < MAXC; ++j) {
        int id = disamb[b * MAXC + j];
        const float* e = embedding + (size_t)id * TRANSE;
        float d = 0.0f;
        for (int k = 0; k < TRANSE; ++k) {
            float df = e[k] - qb[k];
            d += df * df;
        }
        if (d < best) { best = d; bestj = j; }
    }
    int choice = disamb[b * MAXC + bestj];
    bool miss = !((disamb[b * MAXC + 1] == 0) || (best < 0.55f));
    const float* real = (miss || choice == 0) ? qb : (embedding + (size_t)choice * TRANSE);
    const float* fin  = test[b] ? real : (query_ + (size_t)b * TRANSE);
    for (int k = 0; k < TRANSE; ++k) Q16[(size_t)b * KP_H + k] = (_Float16)fin[k];
}

// Squared-score attention over hs (f32) -> ctx [1000,400]. grid=(B,2), block=256.
__global__ void att_ctx_kernel(const float* __restrict__ hs,
                               const float* __restrict__ AQ,
                               float* __restrict__ ctx) {
    __shared__ float red[256];
    __shared__ float s[WIN];
    int b = blockIdx.x;
    int side = blockIdx.y;
    int tid = threadIdx.x;
    int base_out = (2 + side) * BB + b;
    int base_rev = side * BB + b;
    for (int t = 0; t < WIN; ++t) {
        float p = 0.0f;
        if (tid < 2 * HH) {
            int d = tid;
            float v = (d < HH)
                ? hs[((size_t)t * 4 * BB + base_out) * HH + d]
                : hs[((size_t)(WIN - 1 - t) * 4 * BB + base_rev) * HH + (d - HH)];
            p = v * AQ[(size_t)b * 2 * HH + d];
        }
        red[tid] = p;
        __syncthreads();
        for (int off = 128; off > 0; off >>= 1) {
            if (tid < off) red[tid] += red[tid + off];
            __syncthreads();
        }
        if (tid == 0) { float x = red[0]; s[t] = x * x; }
        __syncthreads();
    }
    float ssum = 0.0f;
    for (int t = 0; t < WIN; ++t) ssum += s[t];
    if (tid < 2 * HH) {
        int d = tid;
        float acc = 0.0f;
        for (int t = 0; t < WIN; ++t) {
            float v = (d < HH)
                ? hs[((size_t)t * 4 * BB + base_out) * HH + d]
                : hs[((size_t)(WIN - 1 - t) * 4 * BB + base_rev) * HH + (d - HH)];
            acc += s[t] * v;
        }
        ctx[(size_t)b * 4 * HH + side * 2 * HH + d] = acc / ssum;
    }
}

// xfin16[b][KP_W]: 0..299 entity, 300..699 ctx, pad 0
__global__ void pack_xfin16_kernel(const float* __restrict__ entity,
                                   const float* __restrict__ ctx,
                                   _Float16* __restrict__ xfin) {
    int id = blockIdx.x * blockDim.x + threadIdx.x;
    if (id >= BB * KP_W) return;
    int b = id / KP_W, col = id - b * KP_W;
    float v = 0.0f;
    if (col < WORD)                 v = entity[(size_t)b * WORD + col];
    else if (col < WORD + 4 * HH)   v = ctx[(size_t)b * 4 * HH + (col - WORD)];
    xfin[id] = (_Float16)v;
}

// ---------------------------------------------------------------------------
// Host launcher
// ---------------------------------------------------------------------------
extern "C" void kernel_launch(void* const* d_in, const int* in_sizes, int n_in,
                              void* d_out, int out_size, void* d_ws, size_t ws_size,
                              hipStream_t stream) {
    const float* entity_in = (const float*)d_in[0];
    const float* left_in   = (const float*)d_in[1];
    const float* right_in  = (const float*)d_in[2];
    const float* embedding = (const float*)d_in[3];
    const float* query_    = (const float*)d_in[4];
    const int*   disamb_in = (const int*)d_in[5];
    const unsigned char* test = (const unsigned char*)d_in[6];
    const float* W_lstm    = (const float*)d_in[7];
    const float* b_lstm    = (const float*)d_in[8];
    const float* W_qlstm   = (const float*)d_in[9];
    const float* b_qlstm   = (const float*)d_in[10];
    const float* Wq1       = (const float*)d_in[11];
    const float* Wq2       = (const float*)d_in[12];
    const float* Amat      = (const float*)d_in[13];
    const float* Wfin      = (const float*)d_in[14];
    const float* Tfin      = (const float*)d_in[15];
    float* out = (float*)d_out;

    char* base = (char*)d_ws;
    auto allocF = [&](size_t n) { float* p = (float*)base; base += n * sizeof(float); return p; };
    auto allocH = [&](size_t n) { _Float16* p = (_Float16*)base; base += n * sizeof(_Float16); return p; };

    // f32 buffers
    float* gates_main = allocF((size_t)4 * BB * 4 * HH);   // [4000,400]
    float* gates_q    = allocF((size_t)2 * BB * 4 * HH);   // [2000,400]
    float* hs         = allocF((size_t)WIN * 4 * BB * HH); // [15,4000,100]
    float* c_main     = allocF((size_t)4 * BB * HH);       // zero region start
    float* c_q        = allocF((size_t)2 * BB * HH);
    float* query      = allocF((size_t)BB * TRANSE);
    float* AQ         = allocF((size_t)BB * 2 * HH);
    float* ctx        = allocF((size_t)BB * 4 * HH);
    // f16 buffers
    _Float16* X16     = allocH((size_t)WIN * 4 * BB * KP_X);
    _Float16* BtX_l   = allocH((size_t)4 * HH * KP_X);
    _Float16* BtH_l   = allocH((size_t)4 * HH * KP_H);
    _Float16* BtX_q   = allocH((size_t)4 * HH * KP_X);
    _Float16* BtH_q   = allocH((size_t)4 * HH * KP_H);
    _Float16* BtWq1   = allocH((size_t)MID * KP_Q1);
    _Float16* BtWq2   = allocH((size_t)TRANSE * KP_Q2);
    _Float16* BtA     = allocH((size_t)2 * HH * KP_H);
    _Float16* BtW     = allocH((size_t)TRANSE * KP_W);
    _Float16* BtT     = allocH((size_t)TYPES * KP_T);
    _Float16* xq16    = allocH((size_t)BB * KP_Q1);
    _Float16* xfin16  = allocH((size_t)BB * KP_W);
    // pre-zeroed (pad columns must stay 0) -- keep contiguous for one fill
    _Float16* h16_main = allocH((size_t)4 * BB * KP_H);
    _Float16* h16_q    = allocH((size_t)2 * BB * KP_H);
    _Float16* qmid16   = allocH((size_t)BB * KP_Q2);
    _Float16* Q16      = allocH((size_t)BB * KP_H);
    _Float16* f1_16    = allocH((size_t)BB * KP_T);

    auto gemm = [&](const _Float16* Am, const _Float16* Bt, float* C, _Float16* C16,
                    int M, int N, int Kpad, int ldc, int ldc16, int beta, int act) {
        dim3 grid((N + 63) / 64, (M + 63) / 64);
        size_t smem = ((size_t)64 * Kpad + (size_t)64 * A_STRIDE) * sizeof(_Float16);
        wmma_gemm_kernel<<<grid, 128, smem, stream>>>(Am, Bt, C, C16, M, N, Kpad,
                                                      ldc, ldc16, beta, act);
    };

    // ---- init: zero c state (f32) and padded f16 buffers ----
    {
        int n = (4 * BB + 2 * BB) * HH;  // c_main + c_q contiguous
        fill_zero_f_kernel<<<(n + 255) / 256, 256, 0, stream>>>(c_main, n);
        int m = (4 * BB + 2 * BB) * KP_H + BB * KP_Q2 + BB * KP_H + BB * KP_T; // h16..f1_16
        fill_zero_h_kernel<<<(m + 255) / 256, 256, 0, stream>>>(h16_main, m);
    }

    // ---- precompute f16 inputs + transposed/padded weights ----
    {
        long n = (long)WIN * 4 * BB * KP_X;
        pack_x16_kernel<<<(unsigned)((n + 255) / 256), 256, 0, stream>>>(left_in, right_in, X16);
    }
    auto tcvt = [&](const float* src, _Float16* dst, int K, int N, int Kpad, int row0) {
        int n = N * Kpad;
        transpose_cvt_kernel<<<(n + 255) / 256, 256, 0, stream>>>(src, dst, K, N, Kpad, row0);
    };
    tcvt(W_lstm,  BtX_l, WORD, 4 * HH, KP_X, 0);
    tcvt(W_lstm,  BtH_l, HH,   4 * HH, KP_H, WORD);
    tcvt(W_qlstm, BtX_q, WORD, 4 * HH, KP_X, 0);
    tcvt(W_qlstm, BtH_q, HH,   4 * HH, KP_H, WORD);
    tcvt(Wq1,  BtWq1, WORD + 2 * HH, MID,    KP_Q1, 0);
    tcvt(Wq2,  BtWq2, MID,           TRANSE, KP_Q2, 0);
    tcvt(Wfin, BtW,   WORD + 4 * HH, TRANSE, KP_W,  0);
    tcvt(Tfin, BtT,   TRANSE,        TYPES,  KP_T,  0);
    {   // AQ GEMM's B^T is A itself (row-major), just pad K
        int n = 2 * HH * KP_H;
        pad_cvt_kernel<<<(n + 255) / 256, 256, 0, stream>>>(Amat, BtA, TRANSE, 2 * HH, KP_H);
    }

    // ---------------- LSTM recurrences (15 steps, two LSTMs) ----------------
    for (int t = 0; t < WIN; ++t) {
        const _Float16* Xt = X16 + (size_t)t * 4 * BB * KP_X;
        // main LSTM (4000 rows)
        gemm(Xt,       BtX_l, gates_main, nullptr, 4 * BB, 4 * HH, KP_X, 4 * HH, 0, 0, 0);
        gemm(h16_main, BtH_l, gates_main, nullptr, 4 * BB, 4 * HH, KP_H, 4 * HH, 0, 1, 0);
        {
            int n = 4 * BB * HH;
            lstm_cell_kernel<<<(n + 255) / 256, 256, 0, stream>>>(
                gates_main, b_lstm, c_main, h16_main, hs + (size_t)t * 4 * BB * HH, 4 * BB);
        }
        // query LSTM (first 2000 stacked rows)
        gemm(Xt,     BtX_q, gates_q, nullptr, 2 * BB, 4 * HH, KP_X, 4 * HH, 0, 0, 0);
        gemm(h16_q,  BtH_q, gates_q, nullptr, 2 * BB, 4 * HH, KP_H, 4 * HH, 0, 1, 0);
        {
            int n = 2 * BB * HH;
            lstm_cell_kernel<<<(n + 255) / 256, 256, 0, stream>>>(
                gates_q, b_qlstm, c_q, h16_q, nullptr, 2 * BB);
        }
    }

    // ---------------- query MLP ----------------
    {
        int n = BB * KP_Q1;
        pack_xq16_kernel<<<(n + 255) / 256, 256, 0, stream>>>(entity_in, h16_q, xq16);
    }
    gemm(xq16,   BtWq1, nullptr, qmid16, BB, MID,    KP_Q1, 0,      KP_Q2, 0, 1);
    gemm(qmid16, BtWq2, query,   nullptr, BB, TRANSE, KP_Q2, TRANSE, 0,    0, 1);

    // ---------------- disambiguation + Q select ----------------
    disamb_kernel<<<(BB + 127) / 128, 128, 0, stream>>>(embedding, disamb_in, test,
                                                        query_, query, Q16);

    // ---------------- attention ----------------
    gemm(Q16, BtA, AQ, nullptr, BB, 2 * HH, KP_H, 2 * HH, 0, 0, 0);
    att_ctx_kernel<<<dim3(BB, 2), 256, 0, stream>>>(hs, AQ, ctx);

    // ---------------- final MLP ----------------
    {
        int n = BB * KP_W;
        pack_xfin16_kernel<<<(n + 255) / 256, 256, 0, stream>>>(entity_in, ctx, xfin16);
    }
    gemm(xfin16, BtW, nullptr, f1_16, BB, TRANSE, KP_W, 0,     KP_T, 0, 1);
    gemm(f1_16,  BtT, out,     nullptr, BB, TYPES, KP_T, TYPES, 0,   0, 2);
}
